// GGNN_13357348291351
// MI455X (gfx1250) — matrix operand order repaired and checked by reading, not compiled, not run on hardware
//
#include <hip/hip_runtime.h>
#include <hip/hip_bf16.h>
#include <math.h>

// Problem constants (match reference)
#define HD   512
#define MSGD 128
#define NOE_ 151
#define NOP_ 51
#define NE_  4096
#define NP_  16384

typedef __attribute__((ext_vector_type(16))) __bf16 v16bf;
typedef __attribute__((ext_vector_type(8)))  float  v8f;

__device__ inline unsigned short cvt_bf16_u16(float f) {
  unsigned u = __builtin_bit_cast(unsigned, f);
  return (unsigned short)((u + 0x7FFFu + ((u >> 16) & 1u)) >> 16);  // RNE
}

// GEMM flags (compile-time template parameter)
#define GF_TRA  1   // A is stored K x M (use A^T)
#define GF_TRB  2   // B is stored N x K (use B^T)
#define GF_ACC  4   // C += result (second affine of GRU gates)
#define GF_RELU 8   // fused ReLU

// C(MxN) = act( op(A) @ op(B) + bias [+ C] )
// Tile 64x64, BK=32, 128 threads = 4 wave32; wave w owns output rows [w*16, w*16+16).
// LDS tiles are stored in WMMA *fragment order*: lane L's 16 bf16 elements are
// contiguous, so a fragment load is one aligned 32-byte LDS read (ds_load_b128 x2).
// All four B fragments are loaded before the WMMA chain so waits overlap and the
// 4 WMMAs issue back-to-back. Fully-interior tiles take a block-uniform unguarded
// load path (no per-lane v_cmpx / exec masking).
template <int FLAGS>
__global__ __launch_bounds__(128)
void gemm_wmma_bf16(const float* __restrict__ A, const float* __restrict__ B,
                    const float* __restrict__ bias, float* __restrict__ C,
                    int M, int N, int K, int lda, int ldb, int ldc)
{
  // [tile][lane][elem]; 4 tiles x 32 lanes x 16 bf16 = 4KB each
  __shared__ alignas(32) __bf16 AsF[4][32][16];
  __shared__ alignas(32) __bf16 BsF[4][32][16];

  const int m0   = blockIdx.y * 64;
  const int n0   = blockIdx.x * 64;
  const int tid  = threadIdx.x;
  const int wid  = tid >> 5;
  const int lane = tid & 31;
  const int row16  = lane & 15;
  const int laneHi = lane >> 4;

  const bool mfull = (m0 + 64 <= M);
  const bool nfull = (n0 + 64 <= N);

  v8f acc[4] = {};

  for (int k0 = 0; k0 < K; k0 += 32) {
    // Prefetch next K-tile (emits global_prefetch_b8; speculative, cheap)
    if (k0 + 32 < K && tid == 0) {
      __builtin_prefetch(&A[(FLAGS & GF_TRA) ? (size_t)(k0 + 32) * lda + m0
                                             : (size_t)m0 * lda + (k0 + 32)], 0, 1);
      __builtin_prefetch(&B[(FLAGS & GF_TRB) ? (size_t)n0 * ldb + (k0 + 32)
                                             : (size_t)(k0 + 32) * ldb + n0], 0, 1);
    }
    const bool kfull = (k0 + 32 <= K);

    // ---- stage A (64x32) in fragment order ----
    // A frag: lane L -> row M=L%16; VGPR pair pv -> K base:
    //   lanes 0-15: pv 0..3 -> K 0,2,4,6 ; pv 4..7 -> K 16,18,20,22 ; lanes 16-31: +8
    #pragma unroll
    for (int it = 0; it < 8; ++it) {
      int p = it * 128 + tid;
      int mt = p >> 8, rem = p & 255, l = rem >> 3, pv = rem & 7;
      int m  = mt * 16 + (l & 15);
      int kb = (pv < 4 ? 2 * pv : 16 + 2 * (pv - 4)) + (l >> 4) * 8;
      int gm = m0 + m, gk = k0 + kb;
      float v0, v1;
      if (mfull && kfull) {                       // block-uniform fast path
        if (FLAGS & GF_TRA) {
          v0 = A[(size_t)gk * lda + gm];
          v1 = A[(size_t)(gk + 1) * lda + gm];
        } else {
          v0 = A[(size_t)gm * lda + gk];
          v1 = A[(size_t)gm * lda + gk + 1];
        }
      } else {
        v0 = 0.f; v1 = 0.f;
        if (gm < M) {
          if (FLAGS & GF_TRA) {
            if (gk < K)     v0 = A[(size_t)gk * lda + gm];
            if (gk + 1 < K) v1 = A[(size_t)(gk + 1) * lda + gm];
          } else {
            if (gk < K)     v0 = A[(size_t)gm * lda + gk];
            if (gk + 1 < K) v1 = A[(size_t)gm * lda + gk + 1];
          }
        }
      }
      unsigned pk = (unsigned)cvt_bf16_u16(v0) | ((unsigned)cvt_bf16_u16(v1) << 16);
      *reinterpret_cast<unsigned*>(&AsF[mt][l][pv * 2]) = pk;
    }
    // ---- stage B (32x64) in fragment order ----
    // B frag: lane L -> col N=L%16; VGPR pair pv -> K base (same as A):
    //   lanes 0-15: pv 0..3 -> K 0,2,4,6 ; pv 4..7 -> K 16,18,20,22 ; lanes 16-31: +8
    #pragma unroll
    for (int it = 0; it < 8; ++it) {
      int p = it * 128 + tid;
      int nt = p >> 8, rem = p & 255, l = rem >> 3, pv = rem & 7;
      int n  = nt * 16 + (l & 15);
      int kb = (pv < 4 ? 2 * pv : 16 + 2 * (pv - 4)) + (l >> 4) * 8;
      int gn = n0 + n, gk = k0 + kb;
      float v0, v1;
      if (nfull && kfull) {                       // block-uniform fast path
        if (FLAGS & GF_TRB) {
          v0 = B[(size_t)gn * ldb + gk];
          v1 = B[(size_t)gn * ldb + gk + 1];
        } else {
          v0 = B[(size_t)gk * ldb + gn];
          v1 = B[(size_t)(gk + 1) * ldb + gn];
        }
      } else {
        v0 = 0.f; v1 = 0.f;
        if (gn < N) {
          if (FLAGS & GF_TRB) {
            if (gk < K)     v0 = B[(size_t)gn * ldb + gk];
            if (gk + 1 < K) v1 = B[(size_t)gn * ldb + gk + 1];
          } else {
            if (gk < K)     v0 = B[(size_t)gk * ldb + gn];
            if (gk + 1 < K) v1 = B[(size_t)(gk + 1) * ldb + gn];
          }
        }
      }
      unsigned pk = (unsigned)cvt_bf16_u16(v0) | ((unsigned)cvt_bf16_u16(v1) << 16);
      *reinterpret_cast<unsigned*>(&BsF[nt][l][pv * 2]) = pk;
    }
    __syncthreads();

    // ---- compute: load all fragments first, then 4 back-to-back WMMAs ----
    v16bf af = *reinterpret_cast<const v16bf*>(&AsF[wid][lane][0]);
    v16bf bfr[4];
    #pragma unroll
    for (int nt = 0; nt < 4; ++nt)
      bfr[nt] = *reinterpret_cast<const v16bf*>(&BsF[nt][lane][0]);
    #pragma unroll
    for (int nt = 0; nt < 4; ++nt)
      acc[nt] = __builtin_amdgcn_wmma_f32_16x16x32_bf16(
          false, af, false, bfr[nt], (short)0, acc[nt], false, false);
    __syncthreads();
  }

  // Epilogue. C layout: VGPR r -> M = r + 8*laneHi; N = lane%16.
  #pragma unroll
  for (int nt = 0; nt < 4; ++nt) {
    int gn = n0 + nt * 16 + row16;
    if (gn >= N) continue;
    float bv = bias ? bias[gn] : 0.f;
    #pragma unroll
    for (int r = 0; r < 8; ++r) {
      int gm = m0 + wid * 16 + laneHi * 8 + r;
      if (gm >= M) continue;
      float v = acc[nt][r] + bv;
      if (FLAGS & GF_ACC)  v += C[(size_t)gm * ldc + gn];
      if (FLAGS & GF_RELU) v = fmaxf(v, 0.f);
      C[(size_t)gm * ldc + gn] = v;
    }
  }
}

// ---------------- elementwise / data-movement kernels ----------------

__global__ void k_zero(float* p, int n) {
  int i = blockIdx.x * 256 + threadIdx.x;
  if (i < n) p[i] = 0.f;
}

// nodes = eye @ W + b  ==  W + broadcast(b)
__global__ void k_init_nodes(const float* __restrict__ W, const float* __restrict__ b,
                             float* __restrict__ out, int M, int N) {
  int i = blockIdx.x * 256 + threadIdx.x;
  if (i < M * N) out[i] = W[i] + b[i % N];
}

// dst[q, coloff:coloff+128] = src[rel[q,sel], :]   (pred2{subj,obj} @ ms)
__global__ void k_gather128(const int* __restrict__ rel, const float* __restrict__ src,
                            float* __restrict__ dst, int nP, int sel, int coloff, int ldd) {
  int i = blockIdx.x * 256 + threadIdx.x;
  if (i >= nP * MSGD) return;
  int q = i >> 7, j = i & 127;
  int e = rel[q * 2 + sel];
  dst[(size_t)q * ldd + coloff + j] = src[(size_t)e * MSGD + j];
}

// dst[rel[q,sel], coloff:coloff+128] += src[q, :]  (pred2{subj,obj}^T @ ms)
__global__ void k_scatter128(const int* __restrict__ rel, const float* __restrict__ src,
                             float* __restrict__ dst, int nP, int sel, int coloff, int ldd) {
  int i = blockIdx.x * 256 + threadIdx.x;
  if (i >= nP * MSGD) return;
  int q = i >> 7, j = i & 127;
  int e = rel[q * 2 + sel];
  atomicAdd(&dst[(size_t)e * ldd + coloff + j], src[(size_t)q * MSGD + j]);
}

// rh = sigmoid(rpre) * h
__global__ void k_sigmul(const float* __restrict__ rpre, const float* __restrict__ h,
                         float* __restrict__ rh, int n) {
  int i = blockIdx.x * 256 + threadIdx.x;
  if (i >= n) return;
  float r = 1.f / (1.f + expf(-rpre[i]));
  rh[i] = r * h[i];
}

// h = (1 - sigmoid(zpre)) * h + sigmoid(zpre) * tanh(hpre)
__global__ void k_gruout(const float* __restrict__ zpre, const float* __restrict__ hpre,
                         float* __restrict__ h, int n) {
  int i = blockIdx.x * 256 + threadIdx.x;
  if (i >= n) return;
  float z  = 1.f / (1.f + expf(-zpre[i]));
  float hc = tanhf(hpre[i]);
  h[i] = (1.f - z) * h[i] + z * hc;
}

// ---------------- host orchestration ----------------
//
// d_in flat-leaf order (setup_inputs dict insertion order):
//  0 rel_inds  1 obj_probs  2 obj_fmaps  3 vr
//  4..7    init_{oe,op} W/b
//  8..23   send_{oe,op,ie,ip} {W1,b1,W2,b2}
//  24..39  recv_{oe,op,ie,ip} {W1,b1,W2,b2}
//  40..87  GRU {oe,op,ie,ip} x {eq3,eq4,eq5} x {w,u} x {W,b}
//  88..95  out_{img_pred,ont_pred} {W1,b1,W2,b2}
//  96..99  adj e2e, p2e, e2p, p2p

extern "C" void kernel_launch(void* const* d_in, const int* in_sizes, int n_in,
                              void* d_out, int out_size, void* d_ws, size_t ws_size,
                              hipStream_t stream) {
  (void)in_sizes; (void)n_in; (void)out_size; (void)ws_size;
  auto F = [&](int i) { return (const float*)d_in[i]; };
  const int*   rel       = (const int*)d_in[0];
  const float* obj_probs = F(1);
  const float* obj_fmaps = F(2);
  const float* vr        = F(3);
  const float* e2e = F(96); const float* p2e = F(97);
  const float* e2p = F(98); const float* p2p = F(99);

  // workspace bump allocator
  float* wp = (float*)d_ws;
  auto alloc = [&](size_t nf) { float* p = wp; wp += nf; return p; };
  float* nodes_oe = alloc((size_t)NOE_ * HD);
  float* nodes_op = alloc((size_t)NOP_ * HD);
  float* nodes_ie = alloc((size_t)NE_  * HD);
  float* nodes_ip = alloc((size_t)NP_  * HD);
  float* ms_oe = alloc((size_t)NOE_ * MSGD);
  float* ms_op = alloc((size_t)NOP_ * MSGD);
  float* ms_ie = alloc((size_t)NE_  * MSGD);
  float* ms_ip = alloc((size_t)NP_  * MSGD);
  float* hid   = alloc((size_t)NP_ * 384);        // shared hidden (send 256 / recv up to 1664 cols)
  float* rin_oe = alloc((size_t)NOE_ * 1664);
  float* rin_op = alloc((size_t)NOP_ * 896);
  float* rin_ie = alloc((size_t)NE_  * 384);
  float* rin_ip = alloc((size_t)NP_  * 384);
  float* mbuf = alloc((size_t)NP_ * HD);
  float* tZ = alloc((size_t)NP_ * HD);
  float* tR = alloc((size_t)NP_ * HD);
  float* tH = alloc((size_t)NP_ * HD);
  float* rh = alloc((size_t)NP_ * HD);

  auto gemm = [&](const float* A, const float* B, const float* bias, float* C,
                  int M, int N, int K, int lda, int ldb, int ldc, int flags) {
    dim3 g((N + 63) / 64, (M + 63) / 64), b(128);
    switch (flags) {
      case 0:       gemm_wmma_bf16<0>      <<<g, b, 0, stream>>>(A, B, bias, C, M, N, K, lda, ldb, ldc); break;
      case GF_TRA:  gemm_wmma_bf16<GF_TRA> <<<g, b, 0, stream>>>(A, B, bias, C, M, N, K, lda, ldb, ldc); break;
      case GF_TRB:  gemm_wmma_bf16<GF_TRB> <<<g, b, 0, stream>>>(A, B, bias, C, M, N, K, lda, ldb, ldc); break;
      case GF_ACC:  gemm_wmma_bf16<GF_ACC> <<<g, b, 0, stream>>>(A, B, bias, C, M, N, K, lda, ldb, ldc); break;
      case GF_RELU: gemm_wmma_bf16<GF_RELU><<<g, b, 0, stream>>>(A, B, bias, C, M, N, K, lda, ldb, ldc); break;
      default:      gemm_wmma_bf16<0>      <<<g, b, 0, stream>>>(A, B, bias, C, M, N, K, lda, ldb, ldc); break;
    }
  };
  auto zero = [&](float* p, size_t n) {
    k_zero<<<(unsigned)((n + 255) / 256), 256, 0, stream>>>(p, (int)n);
  };

  // ---- init node states ----
  k_init_nodes<<<(NOE_ * HD + 255) / 256, 256, 0, stream>>>(F(4), F(5), nodes_oe, NOE_, HD);
  k_init_nodes<<<(NOP_ * HD + 255) / 256, 256, 0, stream>>>(F(6), F(7), nodes_op, NOP_, HD);
  hipMemcpyAsync(nodes_ie, obj_fmaps, (size_t)NE_ * HD * sizeof(float),
                 hipMemcpyDeviceToDevice, stream);
  hipMemcpyAsync(nodes_ip, vr, (size_t)NP_ * HD * sizeof(float),
                 hipMemcpyDeviceToDevice, stream);

  const int sbase[4] = {8, 12, 16, 20};     // send params
  const int rbase[4] = {24, 28, 32, 36};    // recv params
  const int gbase[4] = {40, 52, 64, 76};    // gru params
  const int rows_[4] = {NOE_, NOP_, NE_, NP_};
  float* nodes_[4]   = {nodes_oe, nodes_op, nodes_ie, nodes_ip};
  float* ms_[4]      = {ms_oe, ms_op, ms_ie, ms_ip};
  const int dcat_[4] = {1664, 896, 384, 384};
  float* rin_[4]     = {rin_oe, rin_op, rin_ie, rin_ip};

  for (int step = 0; step < 3; ++step) {
    // ---- send MLPs: ms = relu(relu(x@W1+b1)@W2+b2) ----
    for (int s = 0; s < 4; ++s) {
      int b = sbase[s], R = rows_[s];
      gemm(nodes_[s], F(b + 0), F(b + 1), hid,    R, 256, HD, HD, 256, 256, GF_RELU);
      gemm(hid,       F(b + 2), F(b + 3), ms_[s], R, MSGD, 256, 256, MSGD, MSGD, GF_RELU);
    }

    // ---- assemble recv inputs ----
    zero(rin_oe, (size_t)NOE_ * 1664);
    zero(rin_op, (size_t)NOP_ * 896);
    zero(rin_ie, (size_t)NE_ * 384);
    zero(rin_ip, (size_t)NP_ * 384);

    // ontology entities: [ 9x A_t^T@ms_oe | 3x P_t^T@ms_op | obj_probs^T@ms_ie ]
    for (int t = 0; t < 9; ++t)
      gemm(e2e + (size_t)t * NOE_ * NOE_, ms_oe, nullptr, rin_oe + t * MSGD,
           NOE_, MSGD, NOE_, NOE_, MSGD, 1664, GF_TRA);
    for (int t = 0; t < 3; ++t)
      gemm(p2e + (size_t)t * NOP_ * NOE_, ms_op, nullptr, rin_oe + 1152 + t * MSGD,
           NOE_, MSGD, NOP_, NOE_, MSGD, 1664, GF_TRA);
    gemm(obj_probs, ms_ie, nullptr, rin_oe + 1536,
         NOE_, MSGD, NE_, NOE_, MSGD, 1664, GF_TRA);

    // ontology predicates: [ 3x E_t^T@ms_oe | 3x P_t^T@ms_op | zeros ]
    for (int t = 0; t < 3; ++t)
      gemm(e2p + (size_t)t * NOE_ * NOP_, ms_oe, nullptr, rin_op + t * MSGD,
           NOP_, MSGD, NOE_, NOP_, MSGD, 896, GF_TRA);
    for (int t = 0; t < 3; ++t)
      gemm(p2p + (size_t)t * NOP_ * NOP_, ms_op, nullptr, rin_op + 384 + t * MSGD,
           NOP_, MSGD, NOP_, NOP_, MSGD, 896, GF_TRA);

    // image entities: [ scatter(subj) | scatter(obj) | obj_probs@ms_oe ]
    k_scatter128<<<(NP_ * MSGD + 255) / 256, 256, 0, stream>>>(rel, ms_ip, rin_ie, NP_, 0, 0, 384);
    k_scatter128<<<(NP_ * MSGD + 255) / 256, 256, 0, stream>>>(rel, ms_ip, rin_ie, NP_, 1, MSGD, 384);
    gemm(obj_probs, ms_oe, nullptr, rin_ie + 256, NE_, MSGD, NOE_, NOE_, MSGD, 384, 0);

    // image predicates: [ gather(subj) | gather(obj) | zeros ]
    k_gather128<<<(NP_ * MSGD + 255) / 256, 256, 0, stream>>>(rel, ms_ie, rin_ip, NP_, 0, 0, 384);
    k_gather128<<<(NP_ * MSGD + 255) / 256, 256, 0, stream>>>(rel, ms_ie, rin_ip, NP_, 1, MSGD, 384);

    // ---- recv MLPs + GRU, per node set ----
    for (int s = 0; s < 4; ++s) {
      int rb = rbase[s], gb = gbase[s], R = rows_[s], D = dcat_[s];
      gemm(rin_[s], F(rb + 0), F(rb + 1), hid,  R, D, D, D, D, D, GF_RELU);
      gemm(hid,     F(rb + 2), F(rb + 3), mbuf, R, HD, D, D, HD, HD, GF_RELU);

      // z-pre, r-pre: x@W + b + h@U + bu
      gemm(mbuf,      F(gb + 0), F(gb + 1), tZ, R, HD, HD, HD, HD, HD, 0);
      gemm(nodes_[s], F(gb + 2), F(gb + 3), tZ, R, HD, HD, HD, HD, HD, GF_ACC);
      gemm(mbuf,      F(gb + 4), F(gb + 5), tR, R, HD, HD, HD, HD, HD, 0);
      gemm(nodes_[s], F(gb + 6), F(gb + 7), tR, R, HD, HD, HD, HD, HD, GF_ACC);
      k_sigmul<<<(R * HD + 255) / 256, 256, 0, stream>>>(tR, nodes_[s], rh, R * HD);
      gemm(mbuf, F(gb + 8),  F(gb + 9),  tH, R, HD, HD, HD, HD, HD, 0);
      gemm(rh,   F(gb + 10), F(gb + 11), tH, R, HD, HD, HD, HD, HD, GF_ACC);
      k_gruout<<<(R * HD + 255) / 256, 256, 0, stream>>>(tZ, tH, nodes_[s], R * HD);
    }
  }

  // ---- output heads (only the last step's result is returned) ----
  // out_ip = relu(nodes_ip@W1+b1)@W2+b2  (reuse tZ/tR)
  gemm(nodes_ip, F(88), F(89), tZ, NP_, HD, HD, HD, HD, HD, GF_RELU);
  gemm(tZ,       F(90), F(91), tR, NP_, HD, HD, HD, HD, HD, 0);
  // out_op (reuse rh/tH, tiny)
  gemm(nodes_op, F(92), F(93), rh, NOP_, HD, HD, HD, HD, HD, GF_RELU);
  gemm(rh,       F(94), F(95), tH, NOP_, HD, HD, HD, HD, HD, 0);
  // pred_cls = out_ip @ out_op^T   [NP x NOP]
  gemm(tR, tH, nullptr, (float*)d_out, NP_, NOP_, HD, HD, HD, NOP_, GF_TRB);
}
